// RecursiveTransformerBlock_16862041604786
// MI455X (gfx1250) — matrix operand order repaired and checked by reading, not compiled, not run on hardware
//
#include <hip/hip_runtime.h>
#include <hip/hip_bf16.h>

// ---------------------------------------------------------------------------
// Recursive transformer block (soft depth mixture), MI455X / gfx1250, wave32.
// bf16 WMMA (v_wmma_f32_16x16x32_bf16) everywhere, f32 accumulation.
// GEMM: 128x128 block tile, 32x64 per-wave tile (8 WMMA / staging round),
// async global->LDS staging for the A tile (ASYNCcnt path) when available.
// ---------------------------------------------------------------------------

#define BB     8
#define TT     1024
#define CC     1024
#define NHEAD  16
#define HD     64
#define MROWS  (BB * TT)          // 8192
#define EPSV   1e-6f

typedef __attribute__((ext_vector_type(16))) __bf16 v16bf;
typedef __attribute__((ext_vector_type(8)))  __bf16 v8bf;
typedef __attribute__((ext_vector_type(8)))  float  v8f;
typedef int v4i __attribute__((vector_size(16)));   // matches builtin param type
typedef v4i __attribute__((address_space(1)))* gv4i_p;
typedef v4i __attribute__((address_space(3)))* lv4i_p;

#if __has_builtin(__builtin_amdgcn_global_load_async_to_lds_b128) && \
    __has_builtin(__builtin_amdgcn_s_wait_asynccnt)
#define USE_ASYNC_COPY 1
#else
#define USE_ASYNC_COPY 0
#endif

__device__ __forceinline__ v8f wmma_bf16(v16bf a, v16bf b, v8f c) {
  // (neg_a, A, neg_b, B, c_mod, C, reuse_a, reuse_b)
  return __builtin_amdgcn_wmma_f32_16x16x32_bf16(false, a, false, b, (short)0, c,
                                                 false, false);
}

__device__ __forceinline__ v16bf ld_frag(const __bf16* p0, const __bf16* p1) {
  v8bf lo = *(const v8bf*)p0;
  v8bf hi = *(const v8bf*)p1;
  return __builtin_shufflevector(lo, hi, 0, 1, 2, 3, 4, 5, 6, 7, 8, 9, 10, 11,
                                 12, 13, 14, 15);
}

// 16-lane group reductions via ds_swizzle (group-of-32, xor masks 1,2,4,8).
__device__ __forceinline__ float red_max16(float v) {
  v = fmaxf(v, __int_as_float(__builtin_amdgcn_ds_swizzle(__float_as_int(v), 0x041F)));
  v = fmaxf(v, __int_as_float(__builtin_amdgcn_ds_swizzle(__float_as_int(v), 0x081F)));
  v = fmaxf(v, __int_as_float(__builtin_amdgcn_ds_swizzle(__float_as_int(v), 0x101F)));
  v = fmaxf(v, __int_as_float(__builtin_amdgcn_ds_swizzle(__float_as_int(v), 0x201F)));
  return v;
}
__device__ __forceinline__ float red_sum16(float v) {
  v += __int_as_float(__builtin_amdgcn_ds_swizzle(__float_as_int(v), 0x041F));
  v += __int_as_float(__builtin_amdgcn_ds_swizzle(__float_as_int(v), 0x081F));
  v += __int_as_float(__builtin_amdgcn_ds_swizzle(__float_as_int(v), 0x101F));
  v += __int_as_float(__builtin_amdgcn_ds_swizzle(__float_as_int(v), 0x201F));
  return v;
}

// ---------------------------------------------------------------------------
// f32 -> bf16 conversion
// ---------------------------------------------------------------------------
__global__ void cvt_bf16_kernel(const float* __restrict__ src,
                                __bf16* __restrict__ dst, int n) {
  int i = (blockIdx.x * 256 + threadIdx.x) * 4;
  if (i + 3 < n) {
#pragma unroll
    for (int k = 0; k < 4; ++k) dst[i + k] = (__bf16)src[i + k];
  } else {
    for (int k = 0; k < 4 && i + k < n; ++k) dst[i + k] = (__bf16)src[i + k];
  }
}

// ---------------------------------------------------------------------------
// RMSNorm: one block per row (C = 1024, 256 threads, 4 elems/thread) -> bf16
// ---------------------------------------------------------------------------
__global__ void rmsnorm_kernel(const float* __restrict__ x,
                               const float* __restrict__ w,
                               __bf16* __restrict__ out) {
  __shared__ float red[256];
  int row = blockIdx.x;
  int t = threadIdx.x;
  const float* xr = x + (size_t)row * CC;
  float v[4];
  float s = 0.f;
#pragma unroll
  for (int k = 0; k < 4; ++k) {
    v[k] = xr[t + 256 * k];
    s += v[k] * v[k];
  }
  red[t] = s;
  __syncthreads();
#pragma unroll
  for (int off = 128; off > 0; off >>= 1) {
    if (t < off) red[t] += red[t + off];
    __syncthreads();
  }
  float rinv = rsqrtf(red[0] * (1.0f / CC) + EPSV);
  __bf16* o = out + (size_t)row * CC;
#pragma unroll
  for (int k = 0; k < 4; ++k) {
    int col = t + 256 * k;
    o[col] = (__bf16)(v[k] * rinv * w[col]);
  }
}

// ---------------------------------------------------------------------------
// WMMA GEMM: Cacc[M,N] = A[M,K] (bf16 row-major) * W[N,K]^T (bf16 row-major)
// Block tile 128(M) x 128(N), K-step 32, 256 threads = 8 waves (4x2 grid),
// per-wave tile 32(M) x 64(N): 8 WMMAs per staging round.
// Epilogue modes:
//   0: store bf16 to OBF (ldc)
//   1: X[idx] += acc                      (residual attn-out)
//   2: OBF = bf16(gelu_exact(acc))        (fc1)
//   3: X[idx] += acc; OUT (=/+)= p*X[idx] (fc2 + depth mixture)
// ---------------------------------------------------------------------------
#define AS_STRIDE 40   // halves; 80B row (16B aligned)
#define BS_STRIDE 136  // halves; 272B row (16B aligned)

template <int MODE>
__global__ __launch_bounds__(256) void gemm_kernel(
    const __bf16* __restrict__ A, const __bf16* __restrict__ W, int K, int N,
    int ldc, __bf16* __restrict__ OBF, float* __restrict__ X,
    float* __restrict__ OUT, const float* __restrict__ PROBS, int depth,
    int first) {
  __shared__ __bf16 As[128][AS_STRIDE];
  __shared__ __bf16 Bs[32][BS_STRIDE];

  const int tid = threadIdx.x;
  const int wave = tid >> 5;
  const int lane = tid & 31;
  const int hi = lane >> 4;
  const int lr = lane & 15;
  const int m_off = (wave >> 1) * 32;   // 4 M-waves * 32 rows
  const int n_off = (wave & 1) * 64;    // 2 N-waves * 64 cols
  const int gm0 = blockIdx.x * 128;
  const int gn0 = blockIdx.y * 128;

  v8f acc[2][4];
#pragma unroll
  for (int mi = 0; mi < 2; ++mi)
#pragma unroll
    for (int s = 0; s < 4; ++s) acc[mi][s] = (v8f)0.0f;

  for (int k0 = 0; k0 < K; k0 += 32) {
    // --- stage A tile (128x32): 512 16B chunks, 2 per thread ---
#pragma unroll
    for (int i = 0; i < 2; ++i) {
      int c = tid + i * 256;
      int row = c >> 2, ch = c & 3;
      const __bf16* ap = A + (size_t)(gm0 + row) * K + k0 + ch * 8;
#if USE_ASYNC_COPY
      __builtin_amdgcn_global_load_async_to_lds_b128(
          (gv4i_p)(v4i*)(void*)const_cast<__bf16*>(ap),
          (lv4i_p)(v4i*)(void*)&As[row][ch * 8], 0, 0);
#else
      *(v8bf*)&As[row][ch * 8] = *(const v8bf*)ap;
#endif
      if (k0 + 32 < K) __builtin_prefetch(ap + 32, 0, 1);
    }
    // --- stage B tile transposed: Bs[k][n] = W[gn0+n][k0+k] ---
#pragma unroll
    for (int i = 0; i < 2; ++i) {
      int c = tid * 2 + i;             // 0..511
      int bn = c >> 2, bkc = c & 3;    // 128 rows x 4 chunks
      const __bf16* wp = W + (size_t)(gn0 + bn) * K + k0 + bkc * 8;
      v8bf wv = *(const v8bf*)wp;
      if (k0 + 32 < K) __builtin_prefetch(wp + 32, 0, 1);
#pragma unroll
      for (int j = 0; j < 8; ++j) Bs[bkc * 8 + j][bn] = wv[j];
    }
#if USE_ASYNC_COPY
    __builtin_amdgcn_s_wait_asynccnt(0);
#endif
    __syncthreads();

    // --- WMMA: A frag per ISA 16-bit 16x32 interleave, B frag lane->K ---
    v16bf af[2];
#pragma unroll
    for (int mi = 0; mi < 2; ++mi)
      af[mi] = ld_frag(&As[m_off + mi * 16 + lr][8 * hi],
                       &As[m_off + mi * 16 + lr][16 + 8 * hi]);
#pragma unroll
    for (int s = 0; s < 4; ++s) {
      v16bf bf;
#pragma unroll
      for (int e = 0; e < 16; ++e)
        bf[e] = Bs[(e < 8) ? (8 * hi + e) : (16 + 8 * hi + e - 8)][n_off + s * 16 + lr];
#pragma unroll
      for (int mi = 0; mi < 2; ++mi)
        acc[mi][s] = wmma_bf16(af[mi], bf, acc[mi][s]);
    }
    __syncthreads();
  }

  // --- epilogue (C layout: elem j of lane L -> row j+8*hi, col L&15) ---
#pragma unroll
  for (int mi = 0; mi < 2; ++mi) {
#pragma unroll
    for (int s = 0; s < 4; ++s) {
#pragma unroll
      for (int j = 0; j < 8; ++j) {
        int gr = gm0 + m_off + mi * 16 + 8 * hi + j;
        int gc = gn0 + n_off + s * 16 + lr;
        float val = acc[mi][s][j];
        if (MODE == 0) {
          OBF[(size_t)gr * ldc + gc] = (__bf16)val;
        } else if (MODE == 1) {
          size_t idx = (size_t)gr * CC + gc;
          X[idx] += val;
        } else if (MODE == 2) {
          float g = 0.5f * val * (1.0f + erff(val * 0.70710678118654752f));
          OBF[(size_t)gr * ldc + gc] = (__bf16)g;
        } else {  // MODE == 3
          size_t idx = (size_t)gr * CC + gc;
          float xv = X[idx] + val;
          X[idx] = xv;
          float p = PROBS[(size_t)gr * 4 + depth];
          if (first) OUT[idx] = p * xv; else OUT[idx] += p * xv;
        }
      }
    }
  }
}

// ---------------------------------------------------------------------------
// Flash attention over qkv (B,T,3C) bf16. grid=(T/64, B*H), block=128 (4 waves)
// Each wave: 16 queries. K tile (64 keys) staged transposed in LDS.
// ---------------------------------------------------------------------------
__global__ __launch_bounds__(128) void attn_kernel(const __bf16* __restrict__ qkv,
                                                   __bf16* __restrict__ obf) {
  __shared__ __bf16 Kt[64][72];       // [feature][key], 144B rows (16B aligned)
  __shared__ __bf16 Ps[4][16][72];    // per-wave P relayout buffer

  const int bh = blockIdx.y;
  const int b = bh >> 4, h = bh & 15;
  const int qbase = blockIdx.x * 64;
  const int tid = threadIdx.x;
  const int wave = tid >> 5, lane = tid & 31;
  const int hi = lane >> 4, lr = lane & 15;
  const size_t rstride = 3 * CC;  // halves per token row
  const __bf16* base = qkv + (size_t)b * TT * rstride;

  // Q fragments (2 K-steps of 32 features)
  const __bf16* qp = base + (size_t)(qbase + wave * 16 + lr) * rstride + h * HD;
  v16bf qf[2];
#pragma unroll
  for (int ks = 0; ks < 2; ++ks)
    qf[ks] = ld_frag(qp + ks * 32 + 8 * hi, qp + ks * 32 + 16 + 8 * hi);

  float m_s[8], l_s[8];
  v8f oacc[4];
#pragma unroll
  for (int j = 0; j < 8; ++j) { m_s[j] = -3.0e38f; l_s[j] = 0.f; }
#pragma unroll
  for (int n = 0; n < 4; ++n) oacc[n] = (v8f)0.0f;

  for (int kb = 0; kb < TT; kb += 64) {
    // stage K tile transposed: Kt[f][key] = k[kb+key][f]
#pragma unroll
    for (int i = 0; i < 4; ++i) {
      int c = tid * 4 + i;             // 0..511
      int key = c >> 3, fc = c & 7;    // 64 keys x 8 chunks
      v8bf kv = *(const v8bf*)(base + (size_t)(kb + key) * rstride + CC + h * HD + fc * 8);
#pragma unroll
      for (int j = 0; j < 8; ++j) Kt[fc * 8 + j][key] = kv[j];
    }
    __syncthreads();

    // S = Q K^T
    v8f sa[4];
#pragma unroll
    for (int s = 0; s < 4; ++s) sa[s] = (v8f)0.0f;
#pragma unroll
    for (int ks = 0; ks < 2; ++ks) {
#pragma unroll
      for (int s = 0; s < 4; ++s) {
        v16bf bf;
#pragma unroll
        for (int e = 0; e < 16; ++e)
          bf[e] = Kt[ks * 32 + ((e < 8) ? (8 * hi + e) : (16 + 8 * hi + e - 8))][s * 16 + lr];
        sa[s] = wmma_bf16(qf[ks], bf, sa[s]);
      }
    }

    // online softmax (rows j+8*hi; 16-lane group reductions)
#pragma unroll
    for (int j = 0; j < 8; ++j) {
      float sv[4];
      float mx = -3.0e38f;
#pragma unroll
      for (int s = 0; s < 4; ++s) { sv[s] = sa[s][j] * 0.125f; mx = fmaxf(mx, sv[s]); }
      mx = red_max16(mx);
      float mn = fmaxf(m_s[j], mx);
      float corr = __expf(m_s[j] - mn);
      m_s[j] = mn;
      float rs = 0.f;
#pragma unroll
      for (int s = 0; s < 4; ++s) {
        float p = __expf(sv[s] - mn);
        rs += p;
        Ps[wave][8 * hi + j][s * 16 + lr] = (__bf16)p;
      }
      rs = red_sum16(rs);
      l_s[j] = l_s[j] * corr + rs;
#pragma unroll
      for (int n = 0; n < 4; ++n) oacc[n][j] *= corr;
    }

    // O += P V   (P relayout via LDS into A-frag; V already B-operand shaped)
#pragma unroll
    for (int ks = 0; ks < 2; ++ks) {
      v16bf pa = ld_frag(&Ps[wave][lr][ks * 32 + 8 * hi],
                         &Ps[wave][lr][ks * 32 + 16 + 8 * hi]);
      const __bf16* vp = base + (size_t)(kb + ks * 32) * rstride + 2 * CC + h * HD;
#pragma unroll
      for (int n = 0; n < 4; ++n) {
        v16bf vb;
#pragma unroll
        for (int e = 0; e < 16; ++e)
          vb[e] = vp[(size_t)((e < 8) ? (8 * hi + e) : (16 + 8 * hi + e - 8)) * rstride + n * 16 + lr];
        oacc[n] = wmma_bf16(pa, vb, oacc[n]);
      }
    }
    __syncthreads();
  }

  // finalize + store O as bf16 into o buffer (B,T,C)
#pragma unroll
  for (int j = 0; j < 8; ++j) {
    float inv = 1.0f / l_s[j];
    int token = qbase + wave * 16 + 8 * hi + j;
#pragma unroll
    for (int n = 0; n < 4; ++n) {
      float val = oacc[n][j] * inv;
      obf[((size_t)b * TT + token) * CC + h * HD + n * 16 + lr] = (__bf16)val;
    }
  }
}

// ---------------------------------------------------------------------------
// Launcher
// ---------------------------------------------------------------------------
static void launch_gemm(int mode, const __bf16* A, const __bf16* W, int K, int N,
                        int ldc, __bf16* obf, float* x, float* out,
                        const float* probs, int depth, int first,
                        hipStream_t stream) {
  dim3 grid(MROWS / 128, N / 128), blk(256);
  switch (mode) {
    case 0: gemm_kernel<0><<<grid, blk, 0, stream>>>(A, W, K, N, ldc, obf, x, out, probs, depth, first); break;
    case 1: gemm_kernel<1><<<grid, blk, 0, stream>>>(A, W, K, N, ldc, obf, x, out, probs, depth, first); break;
    case 2: gemm_kernel<2><<<grid, blk, 0, stream>>>(A, W, K, N, ldc, obf, x, out, probs, depth, first); break;
    default: gemm_kernel<3><<<grid, blk, 0, stream>>>(A, W, K, N, ldc, obf, x, out, probs, depth, first); break;
  }
}

extern "C" void kernel_launch(void* const* d_in, const int* in_sizes, int n_in,
                              void* d_out, int out_size, void* d_ws, size_t ws_size,
                              hipStream_t stream) {
  const float* x_in  = (const float*)d_in[0];
  // d_in[1] = depths (int64) — unused in soft mode
  const float* probs = (const float*)d_in[2];
  const float* n1w   = (const float*)d_in[3];
  const float* wqkv  = (const float*)d_in[4];
  const float* wout  = (const float*)d_in[5];
  const float* n2w   = (const float*)d_in[6];
  const float* wfc1  = (const float*)d_in[7];
  const float* wfc2  = (const float*)d_in[8];
  float* out = (float*)d_out;

  // workspace carve-up
  char* ws = (char*)d_ws;
  const size_t nQKVW = (size_t)3 * CC * CC;   // 3 145 728
  const size_t nOUTW = (size_t)CC * CC;       // 1 048 576
  const size_t nFC1W = (size_t)4 * CC * CC;   // 4 194 304
  const size_t nFC2W = (size_t)4 * CC * CC;
  const size_t nX    = (size_t)MROWS * CC;    // 8 388 608
  const size_t nQKV  = (size_t)MROWS * 3 * CC;
  const size_t nG    = (size_t)MROWS * 4 * CC;

  __bf16* wqkv_bf = (__bf16*)ws;            ws += nQKVW * 2;
  __bf16* wout_bf = (__bf16*)ws;            ws += nOUTW * 2;
  __bf16* wfc1_bf = (__bf16*)ws;            ws += nFC1W * 2;
  __bf16* wfc2_bf = (__bf16*)ws;            ws += nFC2W * 2;
  float*  x_cur   = (float*)ws;             ws += nX * 4;
  __bf16* h_bf    = (__bf16*)ws;            ws += nX * 2;
  __bf16* qkv_bf  = (__bf16*)ws;            ws += nQKV * 2;
  __bf16* o_bf    = (__bf16*)ws;            ws += nX * 2;
  __bf16* g_bf    = (__bf16*)ws;            ws += nG * 2;

  // one-time conversions + x copy
  cvt_bf16_kernel<<<(int)((nQKVW + 1023) / 1024), 256, 0, stream>>>(wqkv, wqkv_bf, (int)nQKVW);
  cvt_bf16_kernel<<<(int)((nOUTW + 1023) / 1024), 256, 0, stream>>>(wout, wout_bf, (int)nOUTW);
  cvt_bf16_kernel<<<(int)((nFC1W + 1023) / 1024), 256, 0, stream>>>(wfc1, wfc1_bf, (int)nFC1W);
  cvt_bf16_kernel<<<(int)((nFC2W + 1023) / 1024), 256, 0, stream>>>(wfc2, wfc2_bf, (int)nFC2W);
  (void)hipMemcpyAsync(x_cur, x_in, nX * sizeof(float), hipMemcpyDeviceToDevice, stream);

  for (int d = 0; d < 4; ++d) {
    // x = x + attn(rmsnorm(x))
    rmsnorm_kernel<<<MROWS, 256, 0, stream>>>(x_cur, n1w, h_bf);
    launch_gemm(0, h_bf, wqkv_bf, CC, 3 * CC, 3 * CC, qkv_bf, nullptr, nullptr, probs, d, 0, stream);
    attn_kernel<<<dim3(TT / 64, BB * NHEAD), 128, 0, stream>>>(qkv_bf, o_bf);
    launch_gemm(1, o_bf, wout_bf, CC, CC, CC, nullptr, x_cur, nullptr, probs, d, 0, stream);
    // x = x + fc2(gelu(fc1(rmsnorm(x))));  out (=/+)= probs[:,:,d] * x
    rmsnorm_kernel<<<MROWS, 256, 0, stream>>>(x_cur, n2w, h_bf);
    launch_gemm(2, h_bf, wfc1_bf, CC, 4 * CC, 4 * CC, g_bf, nullptr, nullptr, probs, d, 0, stream);
    launch_gemm(3, g_bf, wfc2_bf, 4 * CC, CC, CC, nullptr, x_cur, out, probs, d, (d == 0) ? 1 : 0, stream);
  }
}
